// BIMPM_5669356835283
// MI455X (gfx1250) — compile-verified, matrix operand order checked by reading, not run on hardware
//
#include <hip/hip_runtime.h>
#include <hip/hip_bf16.h>
#include <stdint.h>

// ---------------- problem constants ----------------
#define B_   16
#define S_   96
#define E_   300
#define H_   100
#define G_   400      // 4*H
#define L_   20
#define KE_  320      // E padded to mult of 32
#define KH_  128      // H padded to mult of 32
#define KM_  160      // 8*L (already mult of 32)
#define MT_  (B_*S_)  // 1536
#define EPSF 1e-8f

typedef __bf16 bf16_t;
typedef __attribute__((ext_vector_type(16))) __bf16 v16bf;
typedef __attribute__((ext_vector_type(8)))  __bf16 v8bf;
typedef __attribute__((ext_vector_type(8)))  float  v8f;

// Load a 16x32 bf16 WMMA fragment half for this lane:
// lane r=lane&15 is matrix row; hl=lane>>4 selects K offset +0/+8 (lo) and +16/+24 (hi).
__device__ __forceinline__ v16bf frag_ld(const bf16_t* p) {
  v8bf lo = *(const v8bf*)p;        // K +0..7  (or +8..15 for upper lanes)
  v8bf hi = *(const v8bf*)(p + 16); // K +16..23 (or +24..31)
  v16bf r;
#pragma unroll
  for (int i = 0; i < 8; ++i) { r[i] = lo[i]; r[i + 8] = hi[i]; }
  return r;
}

__device__ __forceinline__ v8f wmma_bf16(v16bf a, v16bf b, v8f c) {
  return __builtin_amdgcn_wmma_f32_16x16x32_bf16(false, a, false, b, (short)0, c,
                                                 false, false);
}

__device__ __forceinline__ float sigmoidf(float x) { return 1.0f / (1.0f + __expf(-x)); }

// order-preserving float<->uint for ds_max_u32-based float max
__device__ __forceinline__ unsigned ford(float f) {
  unsigned u = __float_as_uint(f);
  return (u & 0x80000000u) ? ~u : (u | 0x80000000u);
}
__device__ __forceinline__ float funord(unsigned u) {
  u = (u & 0x80000000u) ? (u & 0x7fffffffu) : ~u;
  return __uint_as_float(u);
}

// ---------------- conversion / gather kernels ----------------
__global__ void k_cvt_bf16(const float* __restrict__ src, bf16_t* __restrict__ dst,
                           int rows, int K, int Kpad) {
  int idx = blockIdx.x * blockDim.x + threadIdx.x;
  if (idx >= rows * Kpad) return;
  int r = idx / Kpad, k = idx - r * Kpad;
  dst[idx] = (k < K) ? (bf16_t)src[(size_t)r * K + k] : (bf16_t)0.0f;
}

__global__ void k_gather(const int* __restrict__ q, const float* __restrict__ emb,
                         bf16_t* __restrict__ X) {
  int idx = blockIdx.x * blockDim.x + threadIdx.x;
  if (idx >= MT_ * KE_) return;
  int r = idx / KE_, k = idx - r * KE_;
  int tok = q[r];
  X[idx] = (k < E_) ? (bf16_t)emb[(size_t)tok * E_ + k] : (bf16_t)0.0f;
}

// ---------------- generic bf16 WMMA GEMM: C[M][N] = A[M][Kp] * B[N][Kp]^T + b1 + b2 ----
__global__ void __launch_bounds__(256)
k_gemm_bf16(const bf16_t* __restrict__ A, const bf16_t* __restrict__ Bm,
            const float* __restrict__ bias1, const float* __restrict__ bias2,
            float* __restrict__ C, int M, int N, int Kpad) {
  int lane = threadIdx.x & 31;
  int wave = threadIdx.x >> 5;
  int tiles_n = N >> 4;
  int tiles   = (M >> 4) * tiles_n;
  int tile = blockIdx.x * (blockDim.x >> 5) + wave;
  if (tile >= tiles) return;
  int tm = tile / tiles_n, tn = tile - tm * tiles_n;
  int hl = lane >> 4, r = lane & 15;
  const bf16_t* ap = A  + (size_t)(tm * 16 + r) * Kpad + hl * 8;
  const bf16_t* bp = Bm + (size_t)(tn * 16 + r) * Kpad + hl * 8;
  v8f acc = {};
  for (int k0 = 0; k0 < Kpad; k0 += 32)
    acc = wmma_bf16(frag_ld(ap + k0), frag_ld(bp + k0), acc);
  int n = tn * 16 + r;
  float badd = bias1[n] + bias2[n];
#pragma unroll
  for (int v = 0; v < 8; ++v) {
    int m = tm * 16 + v + hl * 8;
    C[(size_t)m * N + n] = acc[v] + badd;
  }
}

// ---------------- persistent bidirectional LSTM scan (1 block per direction) --------
__global__ void __launch_bounds__(256)
k_lstm_bidir(const float* __restrict__ xg_f, const float* __restrict__ xg_b,
             const bf16_t* __restrict__ whh_f, const bf16_t* __restrict__ whh_b,
             float* __restrict__ hs_f, float* __restrict__ hs_b) {
  const float*  xg  = (blockIdx.x == 0) ? xg_f  : xg_b;
  const bf16_t* whh = (blockIdx.x == 0) ? whh_f : whh_b;
  float*        hs  = (blockIdx.x == 0) ? hs_f  : hs_b;
  int rev = (int)blockIdx.x;

  __shared__ float  s_c[B_ * H_];
  __shared__ bf16_t s_hb[B_ * KH_];   // bf16 h state, K-padded, WMMA A operand
  __shared__ float  s_g[B_ * G_];     // gate preactivations

  int tid = threadIdx.x;
  for (int i = tid; i < B_ * H_; i += 256) s_c[i] = 0.0f;
  for (int i = tid; i < B_ * KH_; i += 256) s_hb[i] = (bf16_t)0.0f;
  __syncthreads();

  int lane = tid & 31, wave = tid >> 5;
  int hl = lane >> 4, r = lane & 15;

  for (int step = 0; step < S_; ++step) {
    int t = rev ? (S_ - 1 - step) : step;
    // g = h @ Whh^T  (M=16 batch, N=400 gates, K=128)
    for (int tn = wave; tn < G_ / 16; tn += 8) {
      const bf16_t* apl = s_hb + r * KH_ + hl * 8;
      const bf16_t* bp  = whh + (size_t)(tn * 16 + r) * KH_ + hl * 8;
      v8f acc = {};
#pragma unroll
      for (int k0 = 0; k0 < KH_; k0 += 32)
        acc = wmma_bf16(frag_ld(apl + k0), frag_ld(bp + k0), acc);
      int n = tn * 16 + r;
#pragma unroll
      for (int v = 0; v < 8; ++v) {
        int m = v + hl * 8;
        s_g[m * G_ + n] = acc[v] + xg[((size_t)m * S_ + t) * G_ + n];
      }
    }
    __syncthreads();
    // elementwise gates (PyTorch order i,f,g,o)
    for (int i = tid; i < B_ * H_; i += 256) {
      int bb = i / H_, hh = i - bb * H_;
      const float* gb = s_g + bb * G_;
      float ig = sigmoidf(gb[hh]);
      float fg = sigmoidf(gb[H_ + hh]);
      float gg = tanhf(gb[2 * H_ + hh]);
      float og = sigmoidf(gb[3 * H_ + hh]);
      float c = fg * s_c[i] + ig * gg;
      float h = og * tanhf(c);
      s_c[i] = c;
      s_hb[bb * KH_ + hh] = (bf16_t)h;
      hs[((size_t)bb * S_ + t) * H_ + hh] = h;
    }
    __syncthreads();
  }
}

// ---------------- pairwise multi-perspective match (1 block per (b,l)) -------------
__global__ void __launch_bounds__(256)
k_pairwise(const float* __restrict__ p, const float* __restrict__ h,
           const float* __restrict__ w, float* __restrict__ mv_p,
           float* __restrict__ mv_h, int col) {
  int b = blockIdx.x / L_;
  int l = blockIdx.x - b * L_;
  __shared__ bf16_t s_A[S_ * KH_];
  __shared__ bf16_t s_B[S_ * KH_];
  __shared__ float s_na[S_], s_nb[S_];
  __shared__ unsigned s_pm[S_], s_hm[S_];

  int tid = threadIdx.x;
  const float* wl = w + l * H_;
  for (int i = tid; i < S_ * KH_; i += 256) {
    int s = i / KH_, k = i - s * KH_;
    float av = (k < H_) ? wl[k] * p[((size_t)b * S_ + s) * H_ + k] : 0.0f;
    float bv = (k < H_) ? wl[k] * h[((size_t)b * S_ + s) * H_ + k] : 0.0f;
    s_A[i] = (bf16_t)av;
    s_B[i] = (bf16_t)bv;
  }
  for (int s = tid; s < S_; s += 256) {
    float na = 0.0f, nb = 0.0f;
    for (int k = 0; k < H_; ++k) {
      float av = wl[k] * p[((size_t)b * S_ + s) * H_ + k];
      float bv = wl[k] * h[((size_t)b * S_ + s) * H_ + k];
      na += av * av; nb += bv * bv;
    }
    s_na[s] = sqrtf(na); s_nb[s] = sqrtf(nb);
    s_pm[s] = ford(-1e30f); s_hm[s] = ford(-1e30f);
  }
  __syncthreads();

  int lane = tid & 31, wave = tid >> 5;
  int hl = lane >> 4, r = lane & 15;
  for (int tile = wave; tile < 36; tile += 8) {   // 6x6 tiles over 96x96
    int tm = tile / 6, tn = tile - tm * 6;
    const bf16_t* ap = s_A + (tm * 16 + r) * KH_ + hl * 8;
    const bf16_t* bp = s_B + (tn * 16 + r) * KH_ + hl * 8;
    v8f acc = {};
#pragma unroll
    for (int k0 = 0; k0 < KH_; k0 += 32)
      acc = wmma_bf16(frag_ld(ap + k0), frag_ld(bp + k0), acc);
    int n = tn * 16 + r;
#pragma unroll
    for (int v = 0; v < 8; ++v) {
      int m = tm * 16 + v + hl * 8;
      float d = s_na[m] * s_nb[n];
      float c = acc[v] / ((d > EPSF) ? d : EPSF);
      unsigned cu = ford(c);
      atomicMax(&s_pm[m], cu);   // max over s2 -> mv_p
      atomicMax(&s_hm[n], cu);   // max over s1 -> mv_h
    }
  }
  __syncthreads();
  for (int s = tid; s < S_; s += 256) {
    mv_p[((size_t)b * S_ + s) * KM_ + col + l] = funord(s_pm[s]);
    mv_h[((size_t)b * S_ + s) * KM_ + col + l] = funord(s_hm[s]);
  }
}

// ---------------- attention cosine matrix (1 block per b) --------------------------
__global__ void __launch_bounds__(256)
k_attention(const float* __restrict__ p, const float* __restrict__ h,
            float* __restrict__ att) {
  int b = blockIdx.x;
  __shared__ bf16_t s_A[S_ * KH_];
  __shared__ bf16_t s_B[S_ * KH_];
  __shared__ float s_na[S_], s_nb[S_];

  int tid = threadIdx.x;
  for (int i = tid; i < S_ * KH_; i += 256) {
    int s = i / KH_, k = i - s * KH_;
    s_A[i] = (k < H_) ? (bf16_t)p[((size_t)b * S_ + s) * H_ + k] : (bf16_t)0.0f;
    s_B[i] = (k < H_) ? (bf16_t)h[((size_t)b * S_ + s) * H_ + k] : (bf16_t)0.0f;
  }
  for (int s = tid; s < S_; s += 256) {
    float na = 0.0f, nb = 0.0f;
    for (int k = 0; k < H_; ++k) {
      float av = p[((size_t)b * S_ + s) * H_ + k];
      float bv = h[((size_t)b * S_ + s) * H_ + k];
      na += av * av; nb += bv * bv;
    }
    s_na[s] = sqrtf(na); s_nb[s] = sqrtf(nb);
  }
  __syncthreads();

  int lane = tid & 31, wave = tid >> 5;
  int hl = lane >> 4, r = lane & 15;
  for (int tile = wave; tile < 36; tile += 8) {
    int tm = tile / 6, tn = tile - tm * 6;
    const bf16_t* ap = s_A + (tm * 16 + r) * KH_ + hl * 8;
    const bf16_t* bp = s_B + (tn * 16 + r) * KH_ + hl * 8;
    v8f acc = {};
#pragma unroll
    for (int k0 = 0; k0 < KH_; k0 += 32)
      acc = wmma_bf16(frag_ld(ap + k0), frag_ld(bp + k0), acc);
    int n = tn * 16 + r;
#pragma unroll
    for (int v = 0; v < 8; ++v) {
      int m = tm * 16 + v + hl * 8;
      float d = s_na[m] * s_nb[n];
      att[((size_t)b * S_ + m) * S_ + n] = acc[v] / ((d > EPSF) ? d : EPSF);
    }
  }
}

__global__ void k_att_sums(const float* __restrict__ att, float* __restrict__ rowsum,
                           float* __restrict__ colsum) {
  int idx = blockIdx.x * blockDim.x + threadIdx.x;
  if (idx >= B_ * S_) return;
  int b = idx / S_, s = idx - b * S_;
  float rs = 0.0f, cs = 0.0f;
  for (int j = 0; j < S_; ++j) {
    rs += att[((size_t)b * S_ + s) * S_ + j];
    cs += att[((size_t)b * S_ + j) * S_ + s];
  }
  rowsum[idx] = rs; colsum[idx] = cs;
}

// attention-weighted mean and max of vec, reducing over s2 (reduce_second=1) or s1 (=0)
__global__ void k_att_combine(const float* __restrict__ att, const float* __restrict__ vec,
                              const float* __restrict__ sums, float* __restrict__ omean,
                              float* __restrict__ omax, int reduce_second) {
  int idx = blockIdx.x * blockDim.x + threadIdx.x;
  if (idx >= B_ * S_ * H_) return;
  int hh = idx % H_;
  int s  = (idx / H_) % S_;
  int b  = idx / (H_ * S_);
  float acc = 0.0f, mx = -1e30f;
  for (int j = 0; j < S_; ++j) {
    float a = reduce_second ? att[((size_t)b * S_ + s) * S_ + j]
                            : att[((size_t)b * S_ + j) * S_ + s];
    float v = vec[((size_t)b * S_ + j) * H_ + hh];
    float av = a * v;
    acc += av;
    if (av > mx) mx = av;
  }
  float d = sums[b * S_ + s];
  omean[idx] = acc / ((d > EPSF) ? d : EPSF);
  omax[idx]  = mx;
}

// mp_match: cosine(w_l*v1[b,s], w_l*v2[b,(fixed_s|s)]) -> mv[.,.,col+l]
__global__ void k_mp_match(const float* __restrict__ v1, const float* __restrict__ v2,
                           const float* __restrict__ w, float* __restrict__ mv,
                           int col, int v2_fixed_s) {
  int idx = blockIdx.x * blockDim.x + threadIdx.x;
  if (idx >= B_ * S_ * L_) return;
  int l = idx % L_;
  int s = (idx / L_) % S_;
  int b = idx / (L_ * S_);
  const float* wl = w + l * H_;
  const float* a = v1 + ((size_t)b * S_ + s) * H_;
  int s2 = (v2_fixed_s >= 0) ? v2_fixed_s : s;
  const float* c = v2 + ((size_t)b * S_ + s2) * H_;
  float dot = 0.0f, na = 0.0f, nb = 0.0f;
  for (int k = 0; k < H_; ++k) {
    float x = wl[k] * a[k], y = wl[k] * c[k];
    dot += x * y; na += x * x; nb += y * y;
  }
  float d = sqrtf(na) * sqrtf(nb);
  mv[((size_t)b * S_ + s) * KM_ + col + l] = dot / fmaxf(d, EPSF);
}

// ---------------- FC head ----------------
__global__ void k_fc1(const float* __restrict__ apf, const float* __restrict__ apb,
                      const float* __restrict__ ahf, const float* __restrict__ ahb,
                      const float* __restrict__ w, const float* __restrict__ bias,
                      float* __restrict__ out) {
  int idx = blockIdx.x * blockDim.x + threadIdx.x;
  if (idx >= B_ * 2 * H_) return;
  int j = idx % (2 * H_), b = idx / (2 * H_);
  float acc = bias[j];
  for (int k = 0; k < 4 * H_; ++k) {
    float xv;
    if (k < H_)            xv = apf[((size_t)b * S_ + (S_ - 1)) * H_ + k];
    else if (k < 2 * H_)   xv = apb[((size_t)b * S_) * H_ + (k - H_)];
    else if (k < 3 * H_)   xv = ahf[((size_t)b * S_ + (S_ - 1)) * H_ + (k - 2 * H_)];
    else                   xv = ahb[((size_t)b * S_) * H_ + (k - 3 * H_)];
    acc += w[(size_t)j * 4 * H_ + k] * xv;
  }
  out[idx] = tanhf(acc);
}

__global__ void k_fc2_softmax(const float* __restrict__ f1, const float* __restrict__ w,
                              const float* __restrict__ bias, float* __restrict__ out) {
  int b = blockIdx.x * blockDim.x + threadIdx.x;
  if (b >= B_) return;
  float l0 = bias[0], l1 = bias[1];
  for (int k = 0; k < 2 * H_; ++k) {
    float x = f1[b * 2 * H_ + k];
    l0 += w[k] * x;
    l1 += w[2 * H_ + k] * x;
  }
  out[b * 2 + 0] = l0;
  out[b * 2 + 1] = l1;
  float m = fmaxf(l0, l1);
  float e0 = __expf(l0 - m), e1 = __expf(l1 - m);
  float s = e0 + e1;
  out[B_ * 2 + b * 2 + 0] = e0 / s;
  out[B_ * 2 + b * 2 + 1] = e1 / s;
}

// ---------------- host launch ----------------
static inline int cdiv(int a, int b) { return (a + b - 1) / b; }

extern "C" void kernel_launch(void* const* d_in, const int* in_sizes, int n_in,
                              void* d_out, int out_size, void* d_ws, size_t ws_size,
                              hipStream_t stream) {
  (void)in_sizes; (void)n_in; (void)out_size; (void)ws_size;

  const int*   q1     = (const int*)d_in[0];
  const int*   q2     = (const int*)d_in[1];
  const float* emb    = (const float*)d_in[2];
  const float* wih_f  = (const float*)d_in[3];
  const float* whh_f  = (const float*)d_in[4];
  const float* bih_f  = (const float*)d_in[5];
  const float* bhh_f  = (const float*)d_in[6];
  const float* wih_b  = (const float*)d_in[7];
  const float* whh_b  = (const float*)d_in[8];
  const float* bih_b  = (const float*)d_in[9];
  const float* bhh_b  = (const float*)d_in[10];
  const float* mp_w   = (const float*)d_in[11];
  const float* awih_f = (const float*)d_in[12];
  const float* awhh_f = (const float*)d_in[13];
  const float* abih_f = (const float*)d_in[14];
  const float* abhh_f = (const float*)d_in[15];
  const float* awih_b = (const float*)d_in[16];
  const float* awhh_b = (const float*)d_in[17];
  const float* abih_b = (const float*)d_in[18];
  const float* abhh_b = (const float*)d_in[19];
  const float* fc1_w  = (const float*)d_in[20];
  const float* fc1_b  = (const float*)d_in[21];
  const float* fc2_w  = (const float*)d_in[22];
  const float* fc2_b  = (const float*)d_in[23];
  float* out = (float*)d_out;

  // bump allocator over workspace (256B aligned slabs)
  uintptr_t cur = (uintptr_t)d_ws;
  auto alloc = [&](size_t bytes) -> void* {
    uintptr_t p = (cur + 255) & ~(uintptr_t)255;
    cur = p + bytes;
    return (void*)p;
  };

  bf16_t* Xp    = (bf16_t*)alloc((size_t)MT_ * KE_ * 2);
  bf16_t* Xh    = (bf16_t*)alloc((size_t)MT_ * KE_ * 2);
  bf16_t* Wihf  = (bf16_t*)alloc((size_t)G_ * KE_ * 2);
  bf16_t* Wihb  = (bf16_t*)alloc((size_t)G_ * KE_ * 2);
  bf16_t* Whhf  = (bf16_t*)alloc((size_t)G_ * KH_ * 2);
  bf16_t* Whhb  = (bf16_t*)alloc((size_t)G_ * KH_ * 2);
  bf16_t* Waihf = (bf16_t*)alloc((size_t)G_ * KM_ * 2);
  bf16_t* Waihb = (bf16_t*)alloc((size_t)G_ * KM_ * 2);
  bf16_t* Wahhf = (bf16_t*)alloc((size_t)G_ * KH_ * 2);
  bf16_t* Wahhb = (bf16_t*)alloc((size_t)G_ * KH_ * 2);
  float* xg_pf = (float*)alloc((size_t)MT_ * G_ * 4);
  float* xg_pb = (float*)alloc((size_t)MT_ * G_ * 4);
  float* xg_hf = (float*)alloc((size_t)MT_ * G_ * 4);
  float* xg_hb = (float*)alloc((size_t)MT_ * G_ * 4);
  float* p_fw  = (float*)alloc((size_t)MT_ * H_ * 4);
  float* p_bw  = (float*)alloc((size_t)MT_ * H_ * 4);
  float* h_fw  = (float*)alloc((size_t)MT_ * H_ * 4);
  float* h_bw  = (float*)alloc((size_t)MT_ * H_ * 4);
  float* att_fw = (float*)alloc((size_t)B_ * S_ * S_ * 4);
  float* att_bw = (float*)alloc((size_t)B_ * S_ * S_ * 4);
  float* rs_fw = (float*)alloc((size_t)B_ * S_ * 4);
  float* cs_fw = (float*)alloc((size_t)B_ * S_ * 4);
  float* rs_bw = (float*)alloc((size_t)B_ * S_ * 4);
  float* cs_bw = (float*)alloc((size_t)B_ * S_ * 4);
  float* amh_fw = (float*)alloc((size_t)MT_ * H_ * 4);
  float* axh_fw = (float*)alloc((size_t)MT_ * H_ * 4);
  float* amh_bw = (float*)alloc((size_t)MT_ * H_ * 4);
  float* axh_bw = (float*)alloc((size_t)MT_ * H_ * 4);
  float* amp_fw = (float*)alloc((size_t)MT_ * H_ * 4);
  float* axp_fw = (float*)alloc((size_t)MT_ * H_ * 4);
  float* amp_bw = (float*)alloc((size_t)MT_ * H_ * 4);
  float* axp_bw = (float*)alloc((size_t)MT_ * H_ * 4);
  float* mv_p  = (float*)alloc((size_t)MT_ * KM_ * 4);
  float* mv_h  = (float*)alloc((size_t)MT_ * KM_ * 4);
  bf16_t* mvp_bf = (bf16_t*)alloc((size_t)MT_ * KM_ * 2);
  bf16_t* mvh_bf = (bf16_t*)alloc((size_t)MT_ * KM_ * 2);
  float* xg_apf = (float*)alloc((size_t)MT_ * G_ * 4);
  float* xg_apb = (float*)alloc((size_t)MT_ * G_ * 4);
  float* xg_ahf = (float*)alloc((size_t)MT_ * G_ * 4);
  float* xg_ahb = (float*)alloc((size_t)MT_ * G_ * 4);
  float* hs_apf = (float*)alloc((size_t)MT_ * H_ * 4);
  float* hs_apb = (float*)alloc((size_t)MT_ * H_ * 4);
  float* hs_ahf = (float*)alloc((size_t)MT_ * H_ * 4);
  float* hs_ahb = (float*)alloc((size_t)MT_ * H_ * 4);
  float* f1out  = (float*)alloc((size_t)B_ * 2 * H_ * 4);

  // 1) convert weights to padded bf16 [N][Kpad]
  k_cvt_bf16<<<cdiv(G_ * KE_, 256), 256, 0, stream>>>(wih_f,  Wihf,  G_, E_,   KE_);
  k_cvt_bf16<<<cdiv(G_ * KE_, 256), 256, 0, stream>>>(wih_b,  Wihb,  G_, E_,   KE_);
  k_cvt_bf16<<<cdiv(G_ * KH_, 256), 256, 0, stream>>>(whh_f,  Whhf,  G_, H_,   KH_);
  k_cvt_bf16<<<cdiv(G_ * KH_, 256), 256, 0, stream>>>(whh_b,  Whhb,  G_, H_,   KH_);
  k_cvt_bf16<<<cdiv(G_ * KM_, 256), 256, 0, stream>>>(awih_f, Waihf, G_, KM_,  KM_);
  k_cvt_bf16<<<cdiv(G_ * KM_, 256), 256, 0, stream>>>(awih_b, Waihb, G_, KM_,  KM_);
  k_cvt_bf16<<<cdiv(G_ * KH_, 256), 256, 0, stream>>>(awhh_f, Wahhf, G_, H_,   KH_);
  k_cvt_bf16<<<cdiv(G_ * KH_, 256), 256, 0, stream>>>(awhh_b, Wahhb, G_, H_,   KH_);

  // 2) embedding gather to bf16
  k_gather<<<cdiv(MT_ * KE_, 256), 256, 0, stream>>>(q1, emb, Xp);
  k_gather<<<cdiv(MT_ * KE_, 256), 256, 0, stream>>>(q2, emb, Xh);

  // 3) gate precompute GEMMs (M=1536, N=400): xg = X*Wih^T + bih + bhh
  int gblk = cdiv((MT_ / 16) * (G_ / 16), 8);
  k_gemm_bf16<<<gblk, 256, 0, stream>>>(Xp, Wihf, bih_f, bhh_f, xg_pf, MT_, G_, KE_);
  k_gemm_bf16<<<gblk, 256, 0, stream>>>(Xp, Wihb, bih_b, bhh_b, xg_pb, MT_, G_, KE_);
  k_gemm_bf16<<<gblk, 256, 0, stream>>>(Xh, Wihf, bih_f, bhh_f, xg_hf, MT_, G_, KE_);
  k_gemm_bf16<<<gblk, 256, 0, stream>>>(Xh, Wihb, bih_b, bhh_b, xg_hb, MT_, G_, KE_);

  // 4) context BiLSTM scans
  k_lstm_bidir<<<2, 256, 0, stream>>>(xg_pf, xg_pb, Whhf, Whhb, p_fw, p_bw);
  k_lstm_bidir<<<2, 256, 0, stream>>>(xg_hf, xg_hb, Whhf, Whhb, h_fw, h_bw);

  const float* w1 = mp_w + 0 * L_ * H_;
  const float* w2 = mp_w + 1 * L_ * H_;
  const float* w3 = mp_w + 2 * L_ * H_;
  const float* w4 = mp_w + 3 * L_ * H_;
  const float* w5 = mp_w + 4 * L_ * H_;
  const float* w6 = mp_w + 5 * L_ * H_;
  const float* w7 = mp_w + 6 * L_ * H_;
  const float* w8 = mp_w + 7 * L_ * H_;

  // 5) pairwise match max (writes mv cols 20..39 / 100..119)
  k_pairwise<<<B_ * L_, 256, 0, stream>>>(p_fw, h_fw, w3, mv_p, mv_h, 20);
  k_pairwise<<<B_ * L_, 256, 0, stream>>>(p_bw, h_bw, w4, mv_p, mv_h, 100);

  // 6) attention matrices + sums + mean/max combines
  k_attention<<<B_, 256, 0, stream>>>(p_fw, h_fw, att_fw);
  k_attention<<<B_, 256, 0, stream>>>(p_bw, h_bw, att_bw);
  k_att_sums<<<cdiv(B_ * S_, 256), 256, 0, stream>>>(att_fw, rs_fw, cs_fw);
  k_att_sums<<<cdiv(B_ * S_, 256), 256, 0, stream>>>(att_bw, rs_bw, cs_bw);
  int cblk = cdiv(B_ * S_ * H_, 256);
  k_att_combine<<<cblk, 256, 0, stream>>>(att_fw, h_fw, rs_fw, amh_fw, axh_fw, 1);
  k_att_combine<<<cblk, 256, 0, stream>>>(att_bw, h_bw, rs_bw, amh_bw, axh_bw, 1);
  k_att_combine<<<cblk, 256, 0, stream>>>(att_fw, p_fw, cs_fw, amp_fw, axp_fw, 0);
  k_att_combine<<<cblk, 256, 0, stream>>>(att_bw, p_bw, cs_bw, amp_bw, axp_bw, 0);

  // 7) mp_match fills remaining mv columns
  int mblk = cdiv(B_ * S_ * L_, 256);
  k_mp_match<<<mblk, 256, 0, stream>>>(p_fw, h_fw,   w1, mv_p,   0, S_ - 1);
  k_mp_match<<<mblk, 256, 0, stream>>>(p_fw, amh_fw, w5, mv_p,  40, -1);
  k_mp_match<<<mblk, 256, 0, stream>>>(p_fw, axh_fw, w7, mv_p,  60, -1);
  k_mp_match<<<mblk, 256, 0, stream>>>(p_bw, h_bw,   w2, mv_p,  80, 0);
  k_mp_match<<<mblk, 256, 0, stream>>>(p_bw, amh_bw, w6, mv_p, 120, -1);
  k_mp_match<<<mblk, 256, 0, stream>>>(p_bw, axh_bw, w8, mv_p, 140, -1);
  k_mp_match<<<mblk, 256, 0, stream>>>(h_fw, p_fw,   w1, mv_h,   0, S_ - 1);
  k_mp_match<<<mblk, 256, 0, stream>>>(h_fw, amp_fw, w5, mv_h,  40, -1);
  k_mp_match<<<mblk, 256, 0, stream>>>(h_fw, axp_fw, w7, mv_h,  60, -1);
  k_mp_match<<<mblk, 256, 0, stream>>>(h_bw, p_bw,   w2, mv_h,  80, 0);
  k_mp_match<<<mblk, 256, 0, stream>>>(h_bw, amp_bw, w6, mv_h, 120, -1);
  k_mp_match<<<mblk, 256, 0, stream>>>(h_bw, axp_bw, w8, mv_h, 140, -1);

  // 8) aggregation BiLSTM: bf16 convert, gate GEMMs (K=160), scans
  k_cvt_bf16<<<cdiv(MT_ * KM_, 256), 256, 0, stream>>>(mv_p, mvp_bf, MT_, KM_, KM_);
  k_cvt_bf16<<<cdiv(MT_ * KM_, 256), 256, 0, stream>>>(mv_h, mvh_bf, MT_, KM_, KM_);
  k_gemm_bf16<<<gblk, 256, 0, stream>>>(mvp_bf, Waihf, abih_f, abhh_f, xg_apf, MT_, G_, KM_);
  k_gemm_bf16<<<gblk, 256, 0, stream>>>(mvp_bf, Waihb, abih_b, abhh_b, xg_apb, MT_, G_, KM_);
  k_gemm_bf16<<<gblk, 256, 0, stream>>>(mvh_bf, Waihf, abih_f, abhh_f, xg_ahf, MT_, G_, KM_);
  k_gemm_bf16<<<gblk, 256, 0, stream>>>(mvh_bf, Waihb, abih_b, abhh_b, xg_ahb, MT_, G_, KM_);
  k_lstm_bidir<<<2, 256, 0, stream>>>(xg_apf, xg_apb, Wahhf, Wahhb, hs_apf, hs_apb);
  k_lstm_bidir<<<2, 256, 0, stream>>>(xg_ahf, xg_ahb, Wahhf, Wahhb, hs_ahf, hs_ahb);

  // 9) FC head + softmax
  k_fc1<<<cdiv(B_ * 2 * H_, 256), 256, 0, stream>>>(hs_apf, hs_apb, hs_ahf, hs_ahb,
                                                    fc1_w, fc1_b, f1out);
  k_fc2_softmax<<<1, 64, 0, stream>>>(f1out, fc2_w, fc2_b, out);
}
